// DistMultNet_74689481277725
// MI455X (gfx1250) — matrix operand order repaired and checked by reading, not compiled, not run on hardware
//
#include <hip/hip_runtime.h>
#include <math.h>

// ---------------------------------------------------------------------------
// DistMult scoring, restructured as:
//   (1) convert W_head/W_tail/rel_emb f32 -> bf16 staging in workspace
//   (2) node transform GEMM: H = x @ W_head^T + b_head, T = x @ W_tail^T + b_tail
//       (bf16 WMMA 16x16x32, f32 accumulate, results stored bf16 in workspace)
//   (3) per-edge gather + triple-product reduce + sigmoid
// Workspace layout (bytes):
//   [0,                N*256*2)   Hb  (bf16)
//   [N*256*2,        2*N*256*2)   Tb  (bf16)
//   then Whb (65536*2), Wtb (65536*2), Relb (nRel*2)
// Total ~98.2 MiB for N=100000.
// ---------------------------------------------------------------------------

typedef __bf16 bf16;
typedef __bf16 v16bf __attribute__((ext_vector_type(16)));
typedef __bf16 v8bf  __attribute__((ext_vector_type(8)));
typedef float  v8f   __attribute__((ext_vector_type(8)));
typedef float  f4    __attribute__((ext_vector_type(4)));

#define EMB 256   // EMB_DIM == HID_DIM == 256

// ---------------------------------------------------------------- kernel 0 --
__global__ void convert_bf16_kernel(const float* __restrict__ Wh,
                                    const float* __restrict__ Wt,
                                    const float* __restrict__ rel,
                                    bf16* __restrict__ Whb,
                                    bf16* __restrict__ Wtb,
                                    bf16* __restrict__ Relb,
                                    int nW, int nRel) {
  int i = blockIdx.x * blockDim.x + threadIdx.x;
  if (i < nW) {
    Whb[i] = (bf16)Wh[i];
    Wtb[i] = (bf16)Wt[i];
  }
  if (i < nRel) Relb[i] = (bf16)rel[i];
}

// --------------------------------------------------------- frag load helpers --
__device__ __forceinline__ v16bf load_a_frag(const float* p) {
  // one contiguous 64B f32 chunk (16 K values) of one x row -> packed bf16 A-frag
  const f4 f0 = *(const f4*)(p);
  const f4 f1 = *(const f4*)(p + 4);
  const f4 f2 = *(const f4*)(p + 8);
  const f4 f3 = *(const f4*)(p + 12);
  v16bf a;
  a[0]  = (bf16)f0.x; a[1]  = (bf16)f0.y; a[2]  = (bf16)f0.z; a[3]  = (bf16)f0.w;
  a[4]  = (bf16)f1.x; a[5]  = (bf16)f1.y; a[6]  = (bf16)f1.z; a[7]  = (bf16)f1.w;
  a[8]  = (bf16)f2.x; a[9]  = (bf16)f2.y; a[10] = (bf16)f2.z; a[11] = (bf16)f2.w;
  a[12] = (bf16)f3.x; a[13] = (bf16)f3.y; a[14] = (bf16)f3.z; a[15] = (bf16)f3.w;
  return a;
}

// ---------------------------------------------------------------- kernel 1 --
// One block = 16 nodes. 8 waves: waves 0-3 -> W_head N-tiles [0..15],
// waves 4-7 -> W_tail N-tiles [0..15]. Each wave: 4 N-tiles x 8 K-steps of
// v_wmma_f32_16x16x32_bf16, with explicit double-buffering: the A/B fragment
// loads for K-step kt+1 are issued before the 4 WMMAs of K-step kt, so the
// s_wait before each WMMA group covers loads that have been in flight for a
// full WMMA group.
__global__ void __launch_bounds__(256)
node_transform_kernel(const float* __restrict__ x,
                      const float* __restrict__ b_head,
                      const float* __restrict__ b_tail,
                      const bf16* __restrict__ Whb,
                      const bf16* __restrict__ Wtb,
                      bf16* __restrict__ Hb,
                      bf16* __restrict__ Tb,
                      int num_nodes) {
  const int tile = blockIdx.x;          // 16 nodes per tile
  const int tid  = threadIdx.x;
  const int lane = tid & 31;
  const int wave = tid >> 5;            // 0..7
  const int mat  = wave >> 2;           // 0 = head, 1 = tail
  const int nt0  = (wave & 3) * 4;      // first of 4 N-tiles for this wave

  const bf16*  W    = mat ? Wtb : Whb;
  const float* bias = mat ? b_tail : b_head;
  bf16*        Out  = mat ? Tb : Hb;

  const int mrow  = lane & 15;                  // A row / C-D column index
  const int abase = (lane < 16) ? 0 : 16;       // A-frag K sub-offset
  const int kb    = (lane < 16) ? 0 : 16;       // B-frag K sub-offset
  const int rbase = (lane < 16) ? 0 : 8;        // C/D row base

  int nodeA = tile * 16 + mrow;
  if (nodeA >= num_nodes) nodeA = num_nodes - 1;   // clamp (keeps EXEC full)
  const float* xrow = x + (size_t)nodeA * EMB + abase;

  // Per-wave B row pointers (row n of row-major W, fixed K sub-offset kb)
  const bf16* Wp0 = W + (size_t)((nt0 + 0) * 16 + mrow) * EMB + kb;
  const bf16* Wp1 = W + (size_t)((nt0 + 1) * 16 + mrow) * EMB + kb;
  const bf16* Wp2 = W + (size_t)((nt0 + 2) * 16 + mrow) * EMB + kb;
  const bf16* Wp3 = W + (size_t)((nt0 + 3) * 16 + mrow) * EMB + kb;

  // ---- Init accumulators with bias (broadcast over rows; column = N)
  v8f acc0, acc1, acc2, acc3;
  {
    const float b0 = bias[(nt0 + 0) * 16 + mrow];
    const float b1 = bias[(nt0 + 1) * 16 + mrow];
    const float b2 = bias[(nt0 + 2) * 16 + mrow];
    const float b3 = bias[(nt0 + 3) * 16 + mrow];
#pragma unroll
    for (int v = 0; v < 8; ++v) {
      acc0[v] = b0; acc1[v] = b1; acc2[v] = b2; acc3[v] = b3;
    }
  }

  // ---- Prologue: load K-step 0 fragments
  v16bf a_c  = load_a_frag(xrow);
  v16bf b0_c = *(const v16bf*)(Wp0);
  v16bf b1_c = *(const v16bf*)(Wp1);
  v16bf b2_c = *(const v16bf*)(Wp2);
  v16bf b3_c = *(const v16bf*)(Wp3);

  // ---- Main GEMM loop: 8 K-steps x 4 N-tiles, double-buffered
#pragma unroll
  for (int kt = 0; kt < 8; ++kt) {
    v16bf a_n, b0_n, b1_n, b2_n, b3_n;
    if (kt < 7) {
      const int ko = (kt + 1) * 32;
      a_n  = load_a_frag(xrow + ko);
      b0_n = *(const v16bf*)(Wp0 + ko);
      b1_n = *(const v16bf*)(Wp1 + ko);
      b2_n = *(const v16bf*)(Wp2 + ko);
      b3_n = *(const v16bf*)(Wp3 + ko);
    }

    acc0 = __builtin_amdgcn_wmma_f32_16x16x32_bf16(false, a_c, false, b0_c, (short)0, acc0, false, false);
    acc1 = __builtin_amdgcn_wmma_f32_16x16x32_bf16(false, a_c, false, b1_c, (short)0, acc1, false, false);
    acc2 = __builtin_amdgcn_wmma_f32_16x16x32_bf16(false, a_c, false, b2_c, (short)0, acc2, false, false);
    acc3 = __builtin_amdgcn_wmma_f32_16x16x32_bf16(false, a_c, false, b3_c, (short)0, acc3, false, false);

    if (kt < 7) {
      a_c = a_n; b0_c = b0_n; b1_c = b1_n; b2_c = b2_n; b3_c = b3_n;
    }
  }

  // ---- Store D as bf16 (per-v store: lanes 0-15 / 16-31 each cover a
  // contiguous 32B span of one output row -> well coalesced)
#pragma unroll
  for (int v = 0; v < 8; ++v) {
    const int node = tile * 16 + rbase + v;
    if (node < num_nodes) {
      bf16* orow = Out + (size_t)node * EMB + mrow;
      orow[(nt0 + 0) * 16] = (bf16)acc0[v];
      orow[(nt0 + 1) * 16] = (bf16)acc1[v];
      orow[(nt0 + 2) * 16] = (bf16)acc2[v];
      orow[(nt0 + 3) * 16] = (bf16)acc3[v];
    }
  }
}

// ---------------------------------------------------------------- kernel 2 --
// One wave32 per edge. Lane j covers columns [8j, 8j+8): 16B vector gathers of
// H[head], T[tail], rel[type]; local 8-term triple product; xor-tree reduce.
__global__ void __launch_bounds__(256)
score_kernel(const int* __restrict__ heads,
             const int* __restrict__ tails,
             const int* __restrict__ types,
             const bf16* __restrict__ Hb,
             const bf16* __restrict__ Tb,
             const bf16* __restrict__ Relb,
             float* __restrict__ out,
             int num_edges) {
  const int wid  = (blockIdx.x * blockDim.x + threadIdx.x) >> 5;  // edge id
  const int lane = threadIdx.x & 31;
  if (wid >= num_edges) return;

  const int h = heads[wid];
  const int t = tails[wid];
  const int r = types[wid];

  const v8bf hv = *(const v8bf*)(Hb   + (size_t)h * EMB + lane * 8);
  const v8bf tv = *(const v8bf*)(Tb   + (size_t)t * EMB + lane * 8);
  const v8bf rv = *(const v8bf*)(Relb + (size_t)r * EMB + lane * 8);

  float s = 0.f;
#pragma unroll
  for (int k = 0; k < 8; ++k)
    s += (float)hv[k] * (float)rv[k] * (float)tv[k];

  // reduce across 32 lanes
#pragma unroll
  for (int off = 16; off > 0; off >>= 1)
    s += __shfl_xor(s, off, 32);

  if (lane == 0)
    out[wid] = 1.f / (1.f + expf(-s));
}

// ------------------------------------------------------------------- launch --
extern "C" void kernel_launch(void* const* d_in, const int* in_sizes, int n_in,
                              void* d_out, int out_size, void* d_ws, size_t ws_size,
                              hipStream_t stream) {
  // setup_inputs() order:
  //   0:x  1:edge_index  2:edge_type  3:W_head  4:b_head  5:W_tail  6:b_tail  7:rel_emb
  const float* x        = (const float*)d_in[0];
  const int*   eidx     = (const int*)  d_in[1];   // [2, E] flattened
  const int*   etype    = (const int*)  d_in[2];
  const float* W_head   = (const float*)d_in[3];
  const float* b_head   = (const float*)d_in[4];
  const float* W_tail   = (const float*)d_in[5];
  const float* b_tail   = (const float*)d_in[6];
  const float* rel      = (const float*)d_in[7];
  float*       out      = (float*)d_out;

  const int N    = in_sizes[0] / EMB;   // 100000 nodes
  const int E    = in_sizes[1] / 2;     // 250000 edges
  const int nW   = in_sizes[3];         // 65536 (256x256)
  const int nRel = in_sizes[7];         // 128000 (500x256)

  // Workspace carve-up (byte offsets)
  char* wsb = (char*)d_ws;
  const size_t htBytes = (size_t)N * EMB * sizeof(bf16);
  bf16* Hb   = (bf16*)(wsb);
  bf16* Tb   = (bf16*)(wsb + htBytes);
  bf16* Whb  = (bf16*)(wsb + 2 * htBytes);
  bf16* Wtb  = (bf16*)(wsb + 2 * htBytes + (size_t)nW * sizeof(bf16));
  bf16* Relb = (bf16*)(wsb + 2 * htBytes + 2 * (size_t)nW * sizeof(bf16));

  // (1) f32 -> bf16 staging of W_head / W_tail / rel_emb
  {
    const int maxN = (nW > nRel) ? nW : nRel;
    const int blocks = (maxN + 255) / 256;
    convert_bf16_kernel<<<blocks, 256, 0, stream>>>(
        W_head, W_tail, rel, Whb, Wtb, Relb, nW, nRel);
  }

  // (2) node transform GEMM: 16 nodes per block
  {
    const int nTiles = (N + 15) / 16;   // 6250
    node_transform_kernel<<<nTiles, 256, 0, stream>>>(
        x, b_head, b_tail, Whb, Wtb, Hb, Tb, N);
  }

  // (3) per-edge scoring: one wave32 per edge, 8 waves per block
  {
    const int blocks = (E + 7) / 8;     // 31250
    score_kernel<<<blocks, 256, 0, stream>>>(
        eidx, eidx + E, etype, Hb, Tb, Relb, out, E);
  }
}
